// GraphAttnLayer_16561393893395
// MI455X (gfx1250) — matrix operand order repaired and checked by reading, not compiled, not run on hardware
//
#include <hip/hip_runtime.h>
#include <hip/hip_bf16.h>

typedef __attribute__((ext_vector_type(16))) __bf16 v16bf;
typedef __attribute__((ext_vector_type(8)))  float  v8f;

#define N_NODES 4096
#define IN_DIM  128
#define F_DIM   64
#define ALPHA_C 0.2f
#define MASKV   (-9000000000000000.0f)

// LDS B tile: 64 columns x 32 K-halves, column stride 40 halves (80 B) so every
// 16-half fragment (32 B) and every async 16-B segment stays 16-B aligned.
#define BCOL_H   40                 // halves per column
#define BBUF_H   (F_DIM * BCOL_H)   // halves per buffer (2560)

struct B32x { uint4 a, b; };

static __device__ __forceinline__ v16bf loadFrag(const unsigned short* p) {
    const uint4* q = (const uint4*)p;     // p is 16-B aligned by construction
    B32x t; t.a = q[0]; t.b = q[1];
    return __builtin_bit_cast(v16bf, t);
}

// ---------------------------------------------------------------------------
// Kernel 1: h = x @ W (f32), hT in bf16, e2[j] = h[j,:].a2, S1 = sum(a1)
// grid 4096, block 64 (one row per block, one output feature per thread)
// ---------------------------------------------------------------------------
__global__ void __launch_bounds__(64)
k_h(const float* __restrict__ x, const float* __restrict__ wgt,
    const float* __restrict__ att, float* __restrict__ h,
    unsigned short* __restrict__ hbT, float* __restrict__ e2,
    float* __restrict__ s1p)
{
    __shared__ float xs[IN_DIM];
    __shared__ float red[F_DIM];
    const int i = blockIdx.x;
    const int f = threadIdx.x;

    xs[f]      = x[i * IN_DIM + f];
    xs[f + 64] = x[i * IN_DIM + 64 + f];
    __syncthreads();

    float acc = 0.0f;
#pragma unroll
    for (int k = 0; k < IN_DIM; ++k)
        acc = fmaf(xs[k], wgt[k * F_DIM + f], acc);

    h[i * F_DIM + f] = acc;
    __bf16 hv = (__bf16)acc;
    hbT[(size_t)f * N_NODES + i] = __builtin_bit_cast(unsigned short, hv);

    red[f] = acc * att[F_DIM + f];   // a2 = attention[64:]
    __syncthreads();
#pragma unroll
    for (int st = 32; st > 0; st >>= 1) {
        if (f < st) red[f] += red[f + st];
        __syncthreads();
    }
    if (f == 0) e2[i] = red[0];

    if (i == 0 && f == 0) {          // S1 = sum(a1)
        float s = 0.0f;
        for (int t = 0; t < F_DIM; ++t) s += att[t];
        *s1p = s;
    }
}

// ---------------------------------------------------------------------------
// Kernel 2: per-row softmax stats  m[i] = max_j energy, s[i] = sum exp(e-m)
// grid 4096, block 256; thread owns 16 j's (coalesced stride-256 strips)
// ---------------------------------------------------------------------------
__global__ void __launch_bounds__(256)
k_stats(const int* __restrict__ adj, const float* __restrict__ h,
        const float* __restrict__ e2, const float* __restrict__ s1p,
        float* __restrict__ mrow, float* __restrict__ srow)
{
    __shared__ float hs[F_DIM];
    __shared__ float redM[256];
    __shared__ float redS[256];
    const int i   = blockIdx.x;
    const int tid = threadIdx.x;

    if (tid < F_DIM) hs[tid] = h[i * F_DIM + tid] * (*s1p);
    __syncthreads();

    const int* arow = adj + (size_t)i * N_NODES;
    float en[16];
#pragma unroll
    for (int t = 0; t < 16; ++t) {
        const int j = t * 256 + tid;
        float e = hs[j >> 6] + e2[j];
        e = e > 0.0f ? e : ALPHA_C * e;
        en[t] = (arow[j] != 0) ? e : MASKV;
    }
    float m = en[0];
#pragma unroll
    for (int t = 1; t < 16; ++t) m = fmaxf(m, en[t]);
    float s = 0.0f;
#pragma unroll
    for (int t = 0; t < 16; ++t) s += __expf(en[t] - m);

    redM[tid] = m; redS[tid] = s;
    __syncthreads();
#pragma unroll
    for (int st = 128; st > 0; st >>= 1) {
        if (tid < st) {
            const float m2 = fmaxf(redM[tid], redM[tid + st]);
            const float s2 = redS[tid] * __expf(redM[tid] - m2) +
                             redS[tid + st] * __expf(redM[tid + st] - m2);
            redM[tid] = m2; redS[tid] = s2;
        }
        __syncthreads();
    }
    if (tid == 0) { mrow[i] = redM[0]; srow[i] = redS[0]; }
}

// ---------------------------------------------------------------------------
// Kernel 3: out[i,:] = sum_j exp(energy[i,j]-m[i])/s[i] * h[j,:]
// v_wmma_f32_16x16x32_bf16, B tiles streamed to LDS with
// global_load_async_to_lds_b128 (ASYNCcnt) and double buffering.
// grid 64, block 128 (4 waves); wave owns a 16x64 output tile.
// ---------------------------------------------------------------------------
__global__ void __launch_bounds__(128)
k_out(const int* __restrict__ adj, const float* __restrict__ h,
      const unsigned short* __restrict__ hbT, const float* __restrict__ e2,
      const float* __restrict__ s1p, const float* __restrict__ mrow,
      const float* __restrict__ srow, float* __restrict__ out)
{
    __shared__ __attribute__((aligned(128))) unsigned short Bs[2 * BBUF_H];

    const int tid  = threadIdx.x;
    const int wave = tid >> 5;
    const int lane = tid & 31;
    const int hi   = lane >> 4;     // half-wave select
    const int mn   = lane & 15;     // A-row / B-col / D-col within tile
    const int rowBase = (blockIdx.x * 4 + wave) * 16;
    const int irow    = rowBase + mn;           // this lane's A row

    const float S1 = *s1p;
    const float mi = mrow[irow];

    // async-copy mapping: wave covers 16 columns per chunk, 2 instructions,
    // each instruction: 8 columns x 4 lanes x 16 B.
    const int c0  = wave * 16 + (lane >> 2);    // columns c0 and c0+8
    const int seg = lane & 3;                   // 16-B segment within 64-B strip
    const unsigned ldsBase = (unsigned)(size_t)(void*)&Bs[0];

    auto prefetchB = [&](int j0, int buf) {
        const unsigned d0 =
            ldsBase + (unsigned)(buf * (BBUF_H * 2) + c0 * (BCOL_H * 2) + seg * 16);
        const unsigned long long g0 =
            (unsigned long long)(size_t)(hbT + (size_t)c0 * N_NODES + j0 + seg * 8);
        const unsigned d1 = d0 + 8u * (BCOL_H * 2);
        const unsigned long long g1 = g0 + (unsigned long long)8 * N_NODES * 2;
        asm volatile("global_load_async_to_lds_b128 %0, %1, off\n\t"
                     "global_load_async_to_lds_b128 %2, %3, off"
                     :
                     : "v"(d0), "v"(g0), "v"(d1), "v"(g1)
                     : "memory");
    };

    v8f acc0 = {}, acc1 = {}, acc2 = {}, acc3 = {};
    const size_t arowoff = (size_t)irow * N_NODES;

    prefetchB(0, 0);

    for (int j0 = 0; j0 < N_NODES; j0 += 32) {
        const int buf = (j0 >> 5) & 1;

        // current buffer's copy (issued one iteration ago) must be complete
        asm volatile("s_wait_asynccnt 0x0" ::: "memory");
        __syncthreads();

        // overlap: kick off next chunk's B tile into the other buffer
        if (j0 + 32 < N_NODES) prefetchB(j0 + 32, buf ^ 1);

        // prefetch the dominant stream (adj) one chunk ahead
        __builtin_prefetch(adj + arowoff + j0 + 32, 0, 1);

        // ---- build A tile (probabilities) in registers ----
        // A layout: lane(hi,mn) half idx 0..15 -> j = j0 + 16*hi + idx
        const float hterm = h[(size_t)irow * F_DIM + (j0 >> 6)] * S1;
        const int jb = j0 + hi * 16;

        const int4*   ap = (const int4*)(adj + arowoff + jb);
        const int4 A0 = ap[0];
        const int4 A1 = ap[1];
        const int4*   aq = (const int4*)(adj + arowoff + jb + 8);
        const int4 A2 = aq[0];
        const int4 A3 = aq[1];
        const float4* ep = (const float4*)(e2 + jb);
        const float4 E0 = ep[0];
        const float4 E1 = ep[1];
        const float4* eq = (const float4*)(e2 + jb + 8);
        const float4 E2 = eq[0];
        const float4 E3 = eq[1];

        const int   aj[16] = { A0.x, A0.y, A0.z, A0.w, A1.x, A1.y, A1.z, A1.w,
                               A2.x, A2.y, A2.z, A2.w, A3.x, A3.y, A3.z, A3.w };
        const float ee[16] = { E0.x, E0.y, E0.z, E0.w, E1.x, E1.y, E1.z, E1.w,
                               E2.x, E2.y, E2.z, E2.w, E3.x, E3.y, E3.z, E3.w };

        v16bf Av;
#pragma unroll
        for (int t = 0; t < 16; ++t) {
            float e = hterm + ee[t];
            e = e > 0.0f ? e : ALPHA_C * e;
            const float w = (aj[t] != 0) ? __expf(e - mi) : 0.0f;
            Av[t] = (__bf16)w;
        }

        // ---- B fragments from the current LDS buffer ----
        const unsigned short* Bb = Bs + buf * BBUF_H;
        const v16bf B0 = loadFrag(Bb + (0 * 16 + mn) * BCOL_H + hi * 16);
        const v16bf B1 = loadFrag(Bb + (1 * 16 + mn) * BCOL_H + hi * 16);
        const v16bf B2 = loadFrag(Bb + (2 * 16 + mn) * BCOL_H + hi * 16);
        const v16bf B3 = loadFrag(Bb + (3 * 16 + mn) * BCOL_H + hi * 16);

        acc0 = __builtin_amdgcn_wmma_f32_16x16x32_bf16(false, Av, false, B0,
                                                       (short)0, acc0, false, false);
        acc1 = __builtin_amdgcn_wmma_f32_16x16x32_bf16(false, Av, false, B1,
                                                       (short)0, acc1, false, false);
        acc2 = __builtin_amdgcn_wmma_f32_16x16x32_bf16(false, Av, false, B2,
                                                       (short)0, acc2, false, false);
        acc3 = __builtin_amdgcn_wmma_f32_16x16x32_bf16(false, Av, false, B3,
                                                       (short)0, acc3, false, false);
    }

    // ---- epilogue: scale by 1/s[row], store. D layout: row=v+8*hi, col=mn+16n
#pragma unroll
    for (int v = 0; v < 8; ++v) {
        const int r = rowBase + v + 8 * hi;
        const float sinv = 1.0f / srow[r];
        out[(size_t)r * F_DIM +  0 + mn] = acc0[v] * sinv;
        out[(size_t)r * F_DIM + 16 + mn] = acc1[v] * sinv;
        out[(size_t)r * F_DIM + 32 + mn] = acc2[v] * sinv;
        out[(size_t)r * F_DIM + 48 + mn] = acc3[v] * sinv;
    }
}

// ---------------------------------------------------------------------------
extern "C" void kernel_launch(void* const* d_in, const int* in_sizes, int n_in,
                              void* d_out, int out_size, void* d_ws, size_t ws_size,
                              hipStream_t stream)
{
    const float* x   = (const float*)d_in[0];
    const int*   adj = (const int*)  d_in[1];
    const float* wgt = (const float*)d_in[2];
    const float* att = (const float*)d_in[3];
    float* out = (float*)d_out;

    // workspace layout
    float* h    = (float*)d_ws;                 // 4096*64 f32
    float* e2   = h    + N_NODES * F_DIM;       // 4096 f32
    float* mrow = e2   + N_NODES;               // 4096 f32
    float* srow = mrow + N_NODES;               // 4096 f32
    float* s1p  = srow + N_NODES;               // 1 f32 (+pad)
    unsigned short* hbT = (unsigned short*)(s1p + 8);  // 64*4096 bf16 (h^T)

    k_h    <<<N_NODES, 64,  0, stream>>>(x, wgt, att, h, hbT, e2, s1p);
    k_stats<<<N_NODES, 256, 0, stream>>>(adj, h, e2, s1p, mrow, srow);
    k_out  <<<N_NODES / 64, 128, 0, stream>>>(adj, h, hbT, e2, s1p, mrow, srow, out);
}
